// Attention5_16939351016272
// MI455X (gfx1250) — compile-verified, matrix operand order checked by reading, not run on hardware
//
#include <hip/hip_runtime.h>

// ---------------------------------------------------------------------------
// Cross-covariance (channel) attention for MI455X / gfx1250, wave32 + WMMA.
// All big contractions run on v_wmma_f32_16x16x32_bf16 (f32 accumulate).
// Gram stage uses the Tensor Data Mover (tensor_load_to_lds) when the builtin
// is available, with hardware row-padding into conflict-free LDS.
// ---------------------------------------------------------------------------

typedef __attribute__((ext_vector_type(16))) __bf16   v16bf;
typedef __attribute__((ext_vector_type(8)))  float    v8f;
typedef __attribute__((ext_vector_type(4)))  unsigned u32x4;
typedef __attribute__((ext_vector_type(8)))  int      i32x8;
typedef __attribute__((ext_vector_type(4)))  int      i32x4;

union FragB { v16bf v; unsigned u[8]; };

#define BATCH 8
#define DIMC  512
#define MSEQ  8192
#define HEADS 8
#define HC    64
#define MSPLIT 16                  // m-parallelism of the gram stage
#define GCH   (MSEQ / 32 / MSPLIT) // 32-wide K chunks per gram block (=16)

#if defined(__has_builtin)
#if __has_builtin(__builtin_amdgcn_tensor_load_to_lds) && \
    __has_builtin(__builtin_amdgcn_s_wait_tensorcnt)
#define HAVE_TDM 1
#endif
#endif
#ifndef HAVE_TDM
#define HAVE_TDM 0
#endif

#if HAVE_TDM
#warning CDNA5 probe: tensor_load_to_lds path ENABLED in gram stage
#else
#warning CDNA5 probe: tensor_load_to_lds builtin NOT available - DS fallback in gram stage
#endif

// bf16 <-> f32 as raw bits (bf16 stored as ushort everywhere)
__device__ __forceinline__ float bfu_to_f(unsigned s) {
  union { unsigned u; float f; } v;
  v.u = s << 16;
  return v.f;
}
__device__ __forceinline__ unsigned short f_to_bfu(float f) {
  union { float f; unsigned u; } v;
  v.f = f;
  return (unsigned short)((v.u + 0x7FFFu + ((v.u >> 16) & 1u)) >> 16); // RNE
}
__device__ __forceinline__ unsigned pack2(float a, float b) {
  return (unsigned)f_to_bfu(a) | ((unsigned)f_to_bfu(b) << 16);
}

// ISA 7.12.2 16-bit A/B fragment (wave32): r = lane&15 row/col, hf = lane>>4.
// Fragment dwords cover K-pair dword offsets {0,1,2,3, 8,9,10,11} + hf*4
// within the 32-element K chunk.  `base` must be dword-aligned.
__device__ __forceinline__ void load_frag(FragB& f, const unsigned short* base,
                                          int kb0 /* = hf*8 elems */) {
  const unsigned* p = (const unsigned*)(base) + (kb0 >> 1);
#pragma unroll
  for (int j = 0; j < 8; ++j) f.u[j] = p[j < 4 ? j : j + 4];
}

__device__ __forceinline__ v8f vzero8() {
  v8f z;
#pragma unroll
  for (int e = 0; e < 8; ++e) z[e] = 0.f;
  return z;
}

#if HAVE_TDM
// TDM: load a 64(row) x 32(col) bf16 tile (row stride MSEQ elems in memory)
// into LDS with 1 dword of hardware padding per 64B row => LDS row stride 34
// bf16, matching the conflict-free fragment layout.
__device__ __forceinline__ void tdm_load_64x32(const unsigned short* gsrc,
                                               unsigned short* lds_dst) {
  unsigned long long ga = (unsigned long long)(size_t)gsrc;
  unsigned lds = (unsigned)(size_t)lds_dst; // flat addr low 32 bits = LDS offset
  // group0: count=1 | lds_addr | global_addr[56:0] | type=2
  u32x4 g0 = { 1u, lds, (unsigned)ga,
               ((unsigned)(ga >> 32) & 0x01FFFFFFu) | 0x80000000u };
  // group1: wg_mask=0, data_size=2B, pad_enable, pad_interval=16dw, pad=1dw,
  // tensor_dim0=8192, tensor_dim1=64, tile_dim0=32, tile_dim1=64,
  // tensor_dim0_stride=8192
  i32x8 g1 = { 0x00D10000, 0x20000000, 0x00400000, 0x00200000,
               64, 8192, 0, 0 };
  i32x4 gz = { 0, 0, 0, 0 };
#if __clang_major__ >= 23
  i32x8 gz8 = { 0, 0, 0, 0, 0, 0, 0, 0 };
  __builtin_amdgcn_tensor_load_to_lds(g0, g1, gz, gz, gz8, 0);
#else
  __builtin_amdgcn_tensor_load_to_lds(g0, g1, gz, gz, 0);
#endif
}
#endif

// ---------------------------------------------------------------------------
// Generic channel GEMM:  Out[b,o,m] = sum_i W[o,i] * X[b,i,m] + bias[o]
// Block tile 128(o) x 128(m), K chunks of 32. 8 waves: wave = 64(o) x 32(m),
// i.e. 4x2 WMMA tiles. W f32; X f32 or bf16(ushort); Out bf16(ushort) or f32.
// ---------------------------------------------------------------------------
__global__ __launch_bounds__(256, 1) void gemm_wmma_kernel(
    const float* __restrict__ W, const void* __restrict__ Xv,
    const float* __restrict__ bias, void* __restrict__ Outv,
    int Kdim, int Odim, int x_bf16, int out_f32) {
  const int Mdim = MSEQ;
  const int b  = blockIdx.z;
  const int o0 = blockIdx.y * 128;
  const int m0 = blockIdx.x * 128;
  const int tid  = threadIdx.x;
  const int wid  = tid >> 5, lane = tid & 31;
  const int r    = lane & 15, hf = lane >> 4;
  const int wo   = (wid & 1) * 64;
  const int wm   = (wid >> 1) * 32;
  const int kb0  = hf * 8;

  __shared__ __align__(16) unsigned short sW[128 * 34]; // [o][i], stride 17 dw
  __shared__ __align__(16) unsigned short sX[128 * 34]; // [m][i], transposed

  const float*          Xf = (const float*)Xv          + (size_t)b * Kdim * Mdim;
  const unsigned short* Xh = (const unsigned short*)Xv + (size_t)b * Kdim * Mdim;

  v8f acc[4][2];
#pragma unroll
  for (int i = 0; i < 4; ++i)
#pragma unroll
    for (int j = 0; j < 2; ++j) acc[i][j] = vzero8();

  for (int kc = 0; kc < Kdim; kc += 32) {
    // prefetch next X tile (one cacheline per thread)
    if (kc + 32 < Kdim && tid < 128) {
      int row = tid >> 2, seg = tid & 3;
      const char* pf = x_bf16
          ? (const char*)(Xh + (size_t)(kc + 32 + row) * Mdim + m0) + seg * 64
          : (const char*)(Xf + (size_t)(kc + 32 + row) * Mdim + m0) + seg * 128;
      __builtin_prefetch(pf, 0, 3);
    }

    // ---- load phase: global -> registers (clustered wide loads) ----
    float4 wq[4];
#pragma unroll
    for (int it = 0; it < 4; ++it) {
      int idx = it * 256 + tid;      // 1024 quads
      int o = idx >> 3, iq = idx & 7;
      wq[it] = *(const float4*)(W + (size_t)(o0 + o) * Kdim + kc + iq * 4);
    }
    uint2  xb[4];
    float4 xf[4];
    if (x_bf16) {
#pragma unroll
      for (int it = 0; it < 4; ++it) {
        int idx = it * 256 + tid;    // 1024 quads of 4 bf16
        int i = idx >> 5, mq = idx & 31;
        xb[it] = *(const uint2*)(Xh + (size_t)(kc + i) * Mdim + m0 + mq * 4);
      }
    } else {
#pragma unroll
      for (int it = 0; it < 4; ++it) {
        int idx = it * 256 + tid;    // 1024 quads of 4 f32
        int i = idx >> 5, mq = idx & 31;
        xf[it] = *(const float4*)(Xf + (size_t)(kc + i) * Mdim + m0 + mq * 4);
      }
    }

    // ---- store phase: registers -> LDS ----
#pragma unroll
    for (int it = 0; it < 4; ++it) {
      int idx = it * 256 + tid;
      int o = idx >> 3, iq = idx & 7;
      unsigned* dst = (unsigned*)&sW[o * 34 + iq * 4];
      dst[0] = pack2(wq[it].x, wq[it].y);
      dst[1] = pack2(wq[it].z, wq[it].w);
    }
    if (x_bf16) {
#pragma unroll
      for (int it = 0; it < 4; ++it) {
        int idx = it * 256 + tid;
        int i = idx >> 5, mq = idx & 31;
        sX[(mq * 4 + 0) * 34 + i] = (unsigned short)(xb[it].x & 0xFFFFu);
        sX[(mq * 4 + 1) * 34 + i] = (unsigned short)(xb[it].x >> 16);
        sX[(mq * 4 + 2) * 34 + i] = (unsigned short)(xb[it].y & 0xFFFFu);
        sX[(mq * 4 + 3) * 34 + i] = (unsigned short)(xb[it].y >> 16);
      }
    } else {
#pragma unroll
      for (int it = 0; it < 4; ++it) {
        int idx = it * 256 + tid;
        int i = idx >> 5, mq = idx & 31;
        sX[(mq * 4 + 0) * 34 + i] = f_to_bfu(xf[it].x);
        sX[(mq * 4 + 1) * 34 + i] = f_to_bfu(xf[it].y);
        sX[(mq * 4 + 2) * 34 + i] = f_to_bfu(xf[it].z);
        sX[(mq * 4 + 3) * 34 + i] = f_to_bfu(xf[it].w);
      }
    }
    __syncthreads();

    FragB a[4], bb[2];
#pragma unroll
    for (int to = 0; to < 4; ++to)
      load_frag(a[to], sW + (wo + to * 16 + r) * 34, kb0);
#pragma unroll
    for (int tm = 0; tm < 2; ++tm)
      load_frag(bb[tm], sX + (wm + tm * 16 + r) * 34, kb0);

#pragma unroll
    for (int to = 0; to < 4; ++to)
#pragma unroll
      for (int tm = 0; tm < 2; ++tm)
        acc[to][tm] = __builtin_amdgcn_wmma_f32_16x16x32_bf16(
            false, a[to].v, false, bb[tm].v, (short)0, acc[to][tm], false, false);
    __syncthreads();
  }

  // C/D layout: n = lane&15 (m), vgpr j = row j + 8*hf (o)
#pragma unroll
  for (int to = 0; to < 4; ++to) {
#pragma unroll
    for (int tm = 0; tm < 2; ++tm) {
      int m_g = m0 + wm + tm * 16 + r;
#pragma unroll
      for (int j = 0; j < 8; ++j) {
        int o_g = o0 + wo + to * 16 + j + 8 * hf;
        float v = acc[to][tm][j] + bias[o_g];
        size_t off = ((size_t)b * Odim + o_g) * Mdim + m_g;
        if (out_f32) ((float*)Outv)[off] = v;
        else         ((unsigned short*)Outv)[off] = f_to_bfu(v);
      }
    }
  }
}

// ---------------------------------------------------------------------------
// Row norms over M for Q (which=0) and K (which=1, first 512 rows of KV).
// ---------------------------------------------------------------------------
__global__ __launch_bounds__(256) void rownorm_kernel(
    const unsigned short* __restrict__ Qb, const unsigned short* __restrict__ KVb,
    float* __restrict__ nq, float* __restrict__ nk) {
  const int row = blockIdx.x;   // b*512 + c
  const int which = blockIdx.y;
  const int b = row >> 9, c = row & 511;
  const unsigned short* p = which
      ? KVb + ((size_t)b * (2 * DIMC) + c) * MSEQ
      : Qb  + ((size_t)b * DIMC + c) * MSEQ;
  const uint2* p4 = (const uint2*)p;
  float s = 0.f;
#pragma unroll 2
  for (int i = threadIdx.x; i < MSEQ / 4; i += 256) {
    uint2 q = p4[i];
    float a0 = bfu_to_f(q.x & 0xFFFFu), a1 = bfu_to_f(q.x >> 16);
    float a2 = bfu_to_f(q.y & 0xFFFFu), a3 = bfu_to_f(q.y >> 16);
    s += a0 * a0 + a1 * a1 + a2 * a2 + a3 * a3;
  }
  __shared__ float red[256];
  red[threadIdx.x] = s;
  __syncthreads();
  for (int w = 128; w > 0; w >>= 1) {
    if (threadIdx.x < w) red[threadIdx.x] += red[threadIdx.x + w];
    __syncthreads();
  }
  if (threadIdx.x == 0) {
    float n = sqrtf(red[0]);
    (which ? nk : nq)[row] = fmaxf(n, 1e-12f);
  }
}

// ---------------------------------------------------------------------------
// Gram partials: Sp[mb, bh, c, d] = sum_{m in slab} Q[c,m]*K[d,m].
// Grid (MSPLIT, B*H). 8 waves own 2 of the 16 16x16 tiles each.
// TDM double-buffer staging when available, else vectorized LDS staging.
// ---------------------------------------------------------------------------
__global__ __launch_bounds__(256, 1) void gram_kernel(
    const unsigned short* __restrict__ Qb, const unsigned short* __restrict__ KVb,
    float* __restrict__ Sp) {
  const int mb = blockIdx.x;
  const int bh = blockIdx.y, b = bh >> 3, h = bh & 7;
  const int tid = threadIdx.x, wid = tid >> 5, lane = tid & 31;
  const int r = lane & 15, hf = lane >> 4;
  const int tc = wid >> 1, td0 = (wid & 1) * 2;
  const int kb0 = hf * 8;
  const int mbase = mb * (GCH * 32);

  __shared__ __align__(16) unsigned short sQ[2][64 * 34];
  __shared__ __align__(16) unsigned short sK[2][64 * 34];

  const unsigned short* Qrow = Qb  + ((size_t)b * DIMC + h * HC) * MSEQ;
  const unsigned short* Krow = KVb + ((size_t)b * (2 * DIMC) + h * HC) * MSEQ;

  v8f acc[2];
  acc[0] = vzero8();
  acc[1] = vzero8();

#if HAVE_TDM
  if (wid == 0) {
    tdm_load_64x32(Qrow + mbase, &sQ[0][0]);
    tdm_load_64x32(Krow + mbase, &sK[0][0]);
  }
  for (int c = 0; c < GCH; ++c) {
    const int buf = c & 1;
    if (wid == 0) {
      if (c + 1 < GCH) {
        tdm_load_64x32(Qrow + mbase + (c + 1) * 32, &sQ[buf ^ 1][0]);
        tdm_load_64x32(Krow + mbase + (c + 1) * 32, &sK[buf ^ 1][0]);
        __builtin_amdgcn_s_wait_tensorcnt(2); // current pair done, next in flight
      } else {
        __builtin_amdgcn_s_wait_tensorcnt(0);
      }
    }
    __syncthreads();
    FragB a, b0, b1;
    load_frag(a,  &sQ[buf][(tc * 16 + r) * 34], kb0);
    load_frag(b0, &sK[buf][((td0 + 0) * 16 + r) * 34], kb0);
    load_frag(b1, &sK[buf][((td0 + 1) * 16 + r) * 34], kb0);
    acc[0] = __builtin_amdgcn_wmma_f32_16x16x32_bf16(false, a.v, false, b0.v,
                                                     (short)0, acc[0], false, false);
    acc[1] = __builtin_amdgcn_wmma_f32_16x16x32_bf16(false, a.v, false, b1.v,
                                                     (short)0, acc[1], false, false);
    __syncthreads();
  }
#else
  for (int c = 0; c < GCH; ++c) {
    const int buf = c & 1;
    const int mc = mbase + c * 32;
    uint2 qv[2], kv[2];
#pragma unroll
    for (int it = 0; it < 2; ++it) {
      int idx = it * 256 + tid;      // 512 quads per matrix
      int row = idx >> 3, mq = idx & 7;
      qv[it] = *(const uint2*)(Qrow + (size_t)row * MSEQ + mc + mq * 4);
      kv[it] = *(const uint2*)(Krow + (size_t)row * MSEQ + mc + mq * 4);
    }
#pragma unroll
    for (int it = 0; it < 2; ++it) {
      int idx = it * 256 + tid;
      int row = idx >> 3, mq = idx & 7;
      unsigned* dq = (unsigned*)&sQ[buf][row * 34 + mq * 4];
      unsigned* dk = (unsigned*)&sK[buf][row * 34 + mq * 4];
      dq[0] = qv[it].x; dq[1] = qv[it].y;
      dk[0] = kv[it].x; dk[1] = kv[it].y;
    }
    __syncthreads();
    FragB a, b0, b1;
    load_frag(a,  &sQ[buf][(tc * 16 + r) * 34], kb0);
    load_frag(b0, &sK[buf][((td0 + 0) * 16 + r) * 34], kb0);
    load_frag(b1, &sK[buf][((td0 + 1) * 16 + r) * 34], kb0);
    acc[0] = __builtin_amdgcn_wmma_f32_16x16x32_bf16(false, a.v, false, b0.v,
                                                     (short)0, acc[0], false, false);
    acc[1] = __builtin_amdgcn_wmma_f32_16x16x32_bf16(false, a.v, false, b1.v,
                                                     (short)0, acc[1], false, false);
    __syncthreads();
  }
#endif

#pragma unroll
  for (int t = 0; t < 2; ++t) {
    int d = (td0 + t) * 16 + r;
#pragma unroll
    for (int j = 0; j < 8; ++j) {
      int c = tc * 16 + j + 8 * hf;
      Sp[((size_t)mb * (BATCH * HEADS) + bh) * 4096 + c * 64 + d] = acc[t][j];
    }
  }
}

// Deterministic reduction of the MSPLIT gram partials.
__global__ __launch_bounds__(256) void reduceS_kernel(
    const float* __restrict__ Sp, float* __restrict__ S) {
  int idx = blockIdx.x * 256 + threadIdx.x; // 0 .. 262143
  float s = 0.f;
#pragma unroll
  for (int p = 0; p < MSPLIT; ++p)
    s += Sp[(size_t)p * (BATCH * HEADS * 4096) + idx];
  S[idx] = s;
}

// ---------------------------------------------------------------------------
// Softmax over d (64 wide), folding l2-norm scales + temperature.
// ---------------------------------------------------------------------------
__global__ __launch_bounds__(64) void softmax_kernel(
    const float* __restrict__ S, const float* __restrict__ nq,
    const float* __restrict__ nk, const float* __restrict__ temp,
    unsigned short* __restrict__ A) {
  const int bh = blockIdx.x, b = bh >> 3, h = bh & 7;
  const int c = threadIdx.x;
  const float* srow = S + (size_t)bh * 4096 + c * 64;
  const float* nkp = nk + b * DIMC + h * HC;
  const float sc = temp[h] / nq[b * DIMC + h * HC + c];

  float vv[64];
  float mx = -1e30f;
#pragma unroll
  for (int d = 0; d < 64; ++d) {
    vv[d] = srow[d] * sc / nkp[d];
    mx = fmaxf(mx, vv[d]);
  }
  float sum = 0.f;
#pragma unroll
  for (int d = 0; d < 64; ++d) {
    vv[d] = __expf(vv[d] - mx);
    sum += vv[d];
  }
  const float inv = 1.f / sum;
  unsigned short* arow = A + (size_t)bh * 4096 + c * 64;
#pragma unroll
  for (int d = 0; d < 64; ++d) arow[d] = f_to_bfu(vv[d] * inv);
}

// ---------------------------------------------------------------------------
// O1[b, h*64+c, m] = sum_d A[bh,c,d] * V[b, h*64+d, m].  K = 64 (two chunks).
// ---------------------------------------------------------------------------
__global__ __launch_bounds__(256, 1) void apply_kernel(
    const unsigned short* __restrict__ A, const unsigned short* __restrict__ KVb,
    unsigned short* __restrict__ O1) {
  const int bh = blockIdx.y, b = bh >> 3, h = bh & 7;
  const int m0 = blockIdx.x * 128;
  const int tid = threadIdx.x, wid = tid >> 5, lane = tid & 31;
  const int r = lane & 15, hf = lane >> 4;
  const int kb0 = hf * 8;

  __shared__ __align__(16) unsigned short sA[64 * 66];  // [c][d]
  __shared__ __align__(16) unsigned short sV[128 * 66]; // [m][d]

  const unsigned short* Arow = A + (size_t)bh * 4096;
  const unsigned short* Vrow = KVb + ((size_t)b * (2 * DIMC) + DIMC + h * HC) * MSEQ;

  // load phase
  uint2 ar[4], vr[8];
#pragma unroll
  for (int it = 0; it < 4; ++it) {
    int idx = it * 256 + tid;        // 1024 quads
    ar[it] = *(const uint2*)(Arow + idx * 4);
  }
#pragma unroll
  for (int it = 0; it < 8; ++it) {
    int idx = it * 256 + tid;        // 2048 quads
    int d = idx >> 5, mq = idx & 31;
    vr[it] = *(const uint2*)(Vrow + (size_t)d * MSEQ + m0 + mq * 4);
  }
  // store phase
#pragma unroll
  for (int it = 0; it < 4; ++it) {
    int idx = it * 256 + tid;
    int row = idx >> 4, dq = idx & 15;
    unsigned* dst = (unsigned*)&sA[row * 66 + dq * 4];
    dst[0] = ar[it].x; dst[1] = ar[it].y;
  }
#pragma unroll
  for (int it = 0; it < 8; ++it) {
    int idx = it * 256 + tid;
    int d = idx >> 5, mq = idx & 31;
    sV[(mq * 4 + 0) * 66 + d] = (unsigned short)(vr[it].x & 0xFFFFu);
    sV[(mq * 4 + 1) * 66 + d] = (unsigned short)(vr[it].x >> 16);
    sV[(mq * 4 + 2) * 66 + d] = (unsigned short)(vr[it].y & 0xFFFFu);
    sV[(mq * 4 + 3) * 66 + d] = (unsigned short)(vr[it].y >> 16);
  }
  __syncthreads();

  v8f acc[4];
#pragma unroll
  for (int i = 0; i < 4; ++i) acc[i] = vzero8();

#pragma unroll
  for (int kc = 0; kc < 64; kc += 32) {
    FragB bb;
    load_frag(bb, sV + (wid * 16 + r) * 66 + kc, kb0);
#pragma unroll
    for (int tc = 0; tc < 4; ++tc) {
      FragB a;
      load_frag(a, sA + (tc * 16 + r) * 66 + kc, kb0);
      acc[tc] = __builtin_amdgcn_wmma_f32_16x16x32_bf16(
          false, a.v, false, bb.v, (short)0, acc[tc], false, false);
    }
  }

#pragma unroll
  for (int tc = 0; tc < 4; ++tc) {
    int m = m0 + wid * 16 + r;
#pragma unroll
    for (int j = 0; j < 8; ++j) {
      int c = tc * 16 + j + 8 * hf;
      O1[((size_t)b * DIMC + h * HC + c) * MSEQ + m] = f_to_bfu(acc[tc][j]);
    }
  }
}

// ---------------------------------------------------------------------------
// Workspace layout (bytes):
//   KV bf16 [8][1024][8192] @ 0            Q  bf16 @ 134,217,728
//   O1 bf16 @ 201,326,592                  S  f32  @ 268,435,456
//   A  bf16 @ 269,484,032                  nq f32  @ 270,008,320
//   nk f32  @ 270,024,704                  Sp f32  @ 270,041,088 (16 MB)
// ---------------------------------------------------------------------------
extern "C" void kernel_launch(void* const* d_in, const int* in_sizes, int n_in,
                              void* d_out, int out_size, void* d_ws, size_t ws_size,
                              hipStream_t stream) {
  (void)in_sizes; (void)n_in; (void)out_size; (void)ws_size;
  const float* desc = (const float*)d_in[0];
  const float* seg  = (const float*)d_in[1];
  const float* w_kv = (const float*)d_in[2];
  const float* b_kv = (const float*)d_in[3];
  const float* w_q  = (const float*)d_in[4];
  const float* b_q  = (const float*)d_in[5];
  const float* w_po = (const float*)d_in[6];
  const float* b_po = (const float*)d_in[7];
  const float* temp = (const float*)d_in[8];
  float* out = (float*)d_out;

  char* ws = (char*)d_ws;
  unsigned short* KV = (unsigned short*)ws;
  unsigned short* Q  = (unsigned short*)(ws + 134217728ull);
  unsigned short* O1 = (unsigned short*)(ws + 201326592ull);
  float*          S  = (float*)(ws + 268435456ull);
  unsigned short* A  = (unsigned short*)(ws + 269484032ull);
  float*          nq = (float*)(ws + 270008320ull);
  float*          nk = (float*)(ws + 270024704ull);
  float*          Sp = (float*)(ws + 270041088ull);

  const dim3 blk(256);
  gemm_wmma_kernel<<<dim3(MSEQ / 128, 1024 / 128, BATCH), blk, 0, stream>>>(
      w_kv, desc, b_kv, KV, DIMC, 1024, /*x_bf16=*/0, /*out_f32=*/0);
  gemm_wmma_kernel<<<dim3(MSEQ / 128, DIMC / 128, BATCH), blk, 0, stream>>>(
      w_q, seg, b_q, Q, DIMC, DIMC, 0, 0);
  rownorm_kernel<<<dim3(BATCH * DIMC, 2), blk, 0, stream>>>(Q, KV, nq, nk);
  gram_kernel<<<dim3(MSPLIT, BATCH * HEADS), blk, 0, stream>>>(Q, KV, Sp);
  reduceS_kernel<<<dim3(BATCH * HEADS * 4096 / 256), blk, 0, stream>>>(Sp, S);
  softmax_kernel<<<dim3(BATCH * HEADS), dim3(64), 0, stream>>>(S, nq, nk, temp, A);
  apply_kernel<<<dim3(MSEQ / 128, BATCH * HEADS), blk, 0, stream>>>(A, KV, O1);
  gemm_wmma_kernel<<<dim3(MSEQ / 128, DIMC / 128, BATCH), blk, 0, stream>>>(
      w_po, O1, b_po, out, DIMC, DIMC, /*x_bf16=*/1, /*out_f32=*/1);
}
